// Mamba_71932112273563
// MI455X (gfx1250) — matrix operand order repaired and checked by reading, not compiled, not run on hardware
//
#include <hip/hip_runtime.h>
#include <hip/hip_bf16.h>
#include <cstdint>
#include <cstddef>

// Problem constants (match reference)
#define NLAYER 2
#define DMODEL 1024
#define EDIM   2048
#define NSTATE 16
#define KCONV  4
#define RRANK  64
#define BATCH  2
#define LSEQ   1024
#define DBCW   (RRANK + 2 * NSTATE)   // 96

typedef __attribute__((ext_vector_type(16))) _Float16 v16h;
typedef __attribute__((ext_vector_type(8)))  float    v8f;

union HFrag { v16h h; uint32_t u[8]; };

// -------- CDNA5 async global->LDS staging helpers (ASYNCcnt-tracked) -------
__device__ __forceinline__ void async_b128(uint32_t lds_byte, const _Float16* g)
{
    // copies 16B per lane directly into LDS, tracked by ASYNCcnt
    asm volatile("global_load_async_to_lds_b128 %0, %1, off"
                 :: "v"(lds_byte), "v"(g) : "memory");
}
__device__ __forceinline__ void wait_async_le4()
{
    asm volatile("s_wait_asynccnt 0x4" ::: "memory");
}
__device__ __forceinline__ void wait_async_le0()
{
    asm volatile("s_wait_asynccnt 0x0" ::: "memory");
}
__device__ __forceinline__ uint32_t lds_off(const void* p)
{
    // generic (flat) LDS pointer: low 32 bits are the wave-relative LDS offset
    return (uint32_t)(uintptr_t)p;
}

// ---------------- WMMA GEMM: O[M,N] = A[M,K] * W[N,K]^T (+ epilogue) -------
// A, W are f16 (pre-converted). Double-buffered async-to-LDS staging.
// EPI 0: plain f32 store; EPI 1: softplus(val + bias[n]);
// EPI 2: val + res[m,n];  EPI 3: f32 store + f16 shadow store
#define BM 128
#define BN 128
#define BK 32
#define KSTRIDE 40   // f16 elems per LDS row: 80B (16B aligned, bank-staggered)

template<int EPI>
__global__ __launch_bounds__(256)
void gemm_wmma(const _Float16* __restrict__ A, int lda,
               const _Float16* __restrict__ W, int ldb,
               float* __restrict__ O, int ldc,
               const float* __restrict__ X,     // bias (EPI==1) or residual (EPI==2)
               _Float16* __restrict__ O16,      // f16 shadow (EPI==3)
               int N, int Kd)
{
    __shared__ _Float16 As[2][BM * KSTRIDE];
    __shared__ _Float16 Bs[2][BN * KSTRIDE];

    const int tid  = threadIdx.x;
    const int lane = tid & 31;          // wave32
    const int half = lane >> 4;
    const int l15  = lane & 15;
    const int wave = tid >> 5;          // 8 waves
    const int wm   = (wave >> 2) * 64;  // 0,64
    const int wn   = (wave & 3) * 32;   // 0,32,64,96
    const int bm   = blockIdx.y * BM;
    const int bn   = blockIdx.x * BN;

    v8f acc[4][2];
#pragma unroll
    for (int i = 0; i < 4; ++i)
#pragma unroll
        for (int j = 0; j < 2; ++j)
#pragma unroll
            for (int v = 0; v < 8; ++v) acc[i][j][v] = 0.0f;

    // staging assignment: thread t -> tile row r = t>>1, half-row kh = (t&1)*16
    const int r  = tid >> 1;
    const int kh = (tid & 1) << 4;
    const _Float16* aGlob = A + (size_t)(bm + r) * lda + kh;
    const _Float16* wGlob = W + (size_t)(bn + r) * ldb + kh;  // rows padded/valid
    uint32_t aL[2], bL[2];
    aL[0] = lds_off(&As[0][r * KSTRIDE + kh]);
    aL[1] = lds_off(&As[1][r * KSTRIDE + kh]);
    bL[0] = lds_off(&Bs[0][r * KSTRIDE + kh]);
    bL[1] = lds_off(&Bs[1][r * KSTRIDE + kh]);

    auto stage = [&](int buf, int k0) {
        async_b128(aL[buf],      aGlob + k0);
        async_b128(aL[buf] + 16, aGlob + k0 + 8);
        async_b128(bL[buf],      wGlob + k0);
        async_b128(bL[buf] + 16, wGlob + k0 + 8);
    };

    const int KT = Kd / BK;
    stage(0, 0);
    for (int kt = 0; kt < KT; ++kt) {
        const bool more = (kt + 1) < KT;
        if (more) stage((kt + 1) & 1, (kt + 1) * BK);   // prefetch next tile
        if (more) wait_async_le4(); else wait_async_le0(); // oldest group done
        __syncthreads();                                 // tile kt visible

        const _Float16* as = As[kt & 1];
        const _Float16* bs = Bs[kt & 1];
        HFrag a[4], b[2];
#pragma unroll
        for (int i = 0; i < 4; ++i) {
            const _Float16* ap = as + (wm + i * 16 + l15) * KSTRIDE;
#pragma unroll
            for (int v = 0; v < 8; ++v) {
                const int kb = ((v >> 2) << 4) + (half << 3) + ((v & 3) << 1);
                a[i].u[v] = *(const uint32_t*)(ap + kb);
            }
        }
#pragma unroll
        for (int j = 0; j < 2; ++j) {
            const _Float16* bp = bs + (wn + j * 16 + l15) * KSTRIDE;
#pragma unroll
            for (int v = 0; v < 8; ++v) {
                const int kb = ((v >> 2) << 4) + (half << 3) + ((v & 3) << 1);
                b[j].u[v] = *(const uint32_t*)(bp + kb);
            }
        }
#pragma unroll
        for (int i = 0; i < 4; ++i)
#pragma unroll
            for (int j = 0; j < 2; ++j)
                acc[i][j] = __builtin_amdgcn_wmma_f32_16x16x32_f16(
                    false, a[i].h, false, b[j].h, (short)0, acc[i][j],
                    false, false);
        __syncthreads();   // all waves done reading buf[kt&1] before overwrite
    }

    // epilogue + store (C/D layout: VGPR v -> row v + half*8, col = lane&15)
#pragma unroll
    for (int i = 0; i < 4; ++i) {
#pragma unroll
        for (int j = 0; j < 2; ++j) {
            const int coln = bn + wn + j * 16 + l15;
            if (coln < N) {
#pragma unroll
                for (int v = 0; v < 8; ++v) {
                    const int row = bm + wm + i * 16 + half * 8 + v;
                    float val = acc[i][j][v];
                    if (EPI == 1) {
                        const float s = val + X[coln];
                        val = (s > 20.0f) ? s : log1pf(__expf(s));
                    } else if (EPI == 2) {
                        val += X[(size_t)row * ldc + coln];
                    }
                    O[(size_t)row * ldc + coln] = val;
                    if (EPI == 3)
                        O16[(size_t)row * ldc + coln] = (_Float16)val;
                }
            }
        }
    }
}

// ---------------- f32 -> f16 converter (weights; zero-pads tail) -----------
__global__ __launch_bounds__(256)
void cvt_f16_kernel(const float* __restrict__ src, _Float16* __restrict__ dst,
                    int nvalid, int ntotal)
{
    const int i = blockIdx.x * 256 + threadIdx.x;
    if (i < ntotal) dst[i] = (i < nvalid) ? (_Float16)src[i] : (_Float16)0.0f;
}

// ---------------- RMSNorm over D=1024 per (b,l) row -> f16 -----------------
__global__ __launch_bounds__(256)
void rmsnorm_kernel(const float* __restrict__ h, const float* __restrict__ w,
                    _Float16* __restrict__ out)
{
    const int row = blockIdx.x;
    const float* x = h + (size_t)row * DMODEL;
    float s = 0.f;
    for (int c = threadIdx.x; c < DMODEL; c += 256) { const float v = x[c]; s += v * v; }
    __shared__ float red[256];
    red[threadIdx.x] = s;
    __syncthreads();
    for (int o = 128; o > 0; o >>= 1) {
        if (threadIdx.x < o) red[threadIdx.x] += red[threadIdx.x + o];
        __syncthreads();
    }
    const float rs = rsqrtf(red[0] * (1.0f / (float)DMODEL) + 1e-5f);
    for (int c = threadIdx.x; c < DMODEL; c += 256)
        out[(size_t)row * DMODEL + c] = (_Float16)(x[c] * rs * w[c]);
}

// ---------------- Depthwise causal conv (K=4) + SiLU -----------------------
// xz is [B, L, 2*ED]; writes u (f32, for scan) and u_h (f16, for GEMM)
__global__ __launch_bounds__(256)
void conv_silu_kernel(const float* __restrict__ xz, const float* __restrict__ cw,
                      const float* __restrict__ cb, float* __restrict__ u,
                      _Float16* __restrict__ u_h)
{
    const int idx = blockIdx.x * 256 + threadIdx.x;   // (b*L + l)*ED + e
    const int e = idx & (EDIM - 1);
    const int l = (idx >> 11) & (LSEQ - 1);
    const int b = idx >> 21;
    float acc = cb[e];
    const float w0 = cw[e * 4 + 0], w1 = cw[e * 4 + 1];
    const float w2 = cw[e * 4 + 2], w3 = cw[e * 4 + 3];
    const float* base = xz + (size_t)b * LSEQ * (2 * EDIM) + e;
    if (l >= 3) acc += w0 * base[(size_t)(l - 3) * (2 * EDIM)];
    if (l >= 2) acc += w1 * base[(size_t)(l - 2) * (2 * EDIM)];
    if (l >= 1) acc += w2 * base[(size_t)(l - 1) * (2 * EDIM)];
    acc += w3 * base[(size_t)l * (2 * EDIM)];
    const float sv = acc / (1.0f + __expf(-acc));     // silu
    u[idx] = sv;
    u_h[idx] = (_Float16)sv;
}

// ---------------- Selective scan + D*u + silu(z) gate -> f16 ---------------
__global__ __launch_bounds__(256)
void scan_kernel(const float* __restrict__ delta, const float* __restrict__ u,
                 const float* __restrict__ xz, const float* __restrict__ dbc,
                 const float* __restrict__ A_log, const float* __restrict__ Dp,
                 _Float16* __restrict__ yout)
{
    const int b = blockIdx.x >> 3;                       // ED/256 = 8 chunks per b
    const int e = ((blockIdx.x & 7) << 8) + threadIdx.x; // 0..2047
    float Av[NSTATE], hst[NSTATE];
#pragma unroll
    for (int n = 0; n < NSTATE; ++n) {
        Av[n] = -__expf(A_log[e * NSTATE + n]);
        hst[n] = 0.f;
    }
    const float dpe = Dp[e];
    const float* drow = dbc + (size_t)b * LSEQ * DBCW;
    for (int l = 0; l < LSEQ; ++l) {
        const size_t idx = ((size_t)b * LSEQ + l) * EDIM + e;
        // prefetch 8 rows ahead (stays inside workspace; speculative)
        if (l + 8 < LSEQ) {
            __builtin_prefetch(delta + idx + 8 * EDIM, 0, 0);
            __builtin_prefetch(u + idx + 8 * EDIM, 0, 0);
        }
        const float dv = delta[idx];
        const float uv = u[idx];
        const float zv = xz[((size_t)b * LSEQ + l) * (2 * EDIM) + EDIM + e];
        const float* Brow = drow + l * DBCW + RRANK;     // B then C (16 + 16)
        const float dbu = dv * uv;
        float yv = 0.f;
#pragma unroll
        for (int n = 0; n < NSTATE; ++n) {
            const float dA = __expf(dv * Av[n]);
            hst[n] = dA * hst[n] + dbu * Brow[n];
            yv += hst[n] * Brow[NSTATE + n];
        }
        yv += dpe * uv;
        const float sz = zv / (1.0f + __expf(-zv));
        yout[idx] = (_Float16)(yv * sz);
    }
}

// ---------------- Host orchestration ---------------------------------------
extern "C" void kernel_launch(void* const* d_in, const int* in_sizes, int n_in,
                              void* d_out, int out_size, void* d_ws, size_t ws_size,
                              hipStream_t stream)
{
    (void)in_sizes; (void)n_in; (void)out_size; (void)ws_size;

    const float* x      = (const float*)d_in[0];
    const float* in_w   = (const float*)d_in[1];
    const float* conv_w = (const float*)d_in[2];
    const float* conv_b = (const float*)d_in[3];
    const float* xp_w   = (const float*)d_in[4];
    const float* dt_w   = (const float*)d_in[5];
    const float* dt_b   = (const float*)d_in[6];
    const float* A_log  = (const float*)d_in[7];
    const float* D_p    = (const float*)d_in[8];
    const float* out_w  = (const float*)d_in[9];
    const float* norm_w = (const float*)d_in[10];
    float* h = (float*)d_out;                 // residual stream lives in d_out

    // workspace carve-out (bytes, 256-aligned)
    char* ws = (char*)d_ws;
    size_t off = 0;
    auto carve = [&](size_t bytes) { void* p = ws + off; off += (bytes + 255) & ~(size_t)255; return p; };
    float*    xz      = (float*)   carve((size_t)BATCH * LSEQ * 2 * EDIM * 4); // 32 MB
    float*    u       = (float*)   carve((size_t)BATCH * LSEQ * EDIM * 4);     // 16 MB
    float*    dbc     = (float*)   carve((size_t)BATCH * LSEQ * DBCW * 4);     // .75 MB
    float*    delta   = (float*)   carve((size_t)BATCH * LSEQ * EDIM * 4);     // 16 MB
    _Float16* xn_h    = (_Float16*)carve((size_t)BATCH * LSEQ * DMODEL * 2);   //  4 MB
    _Float16* u_h     = (_Float16*)carve((size_t)BATCH * LSEQ * EDIM * 2);     //  8 MB
    _Float16* dbc_h   = (_Float16*)carve((size_t)BATCH * LSEQ * DBCW * 2);     // .4 MB
    _Float16* y_h     = (_Float16*)carve((size_t)BATCH * LSEQ * EDIM * 2);     //  8 MB
    _Float16* in_w_h  = (_Float16*)carve((size_t)2 * EDIM * DMODEL * 2);       //  8 MB
    _Float16* xp_w_h  = (_Float16*)carve((size_t)BN * EDIM * 2);               // .5 MB (padded rows)
    _Float16* dt_w_h  = (_Float16*)carve((size_t)EDIM * RRANK * 2);            // .25 MB
    _Float16* out_w_h = (_Float16*)carve((size_t)DMODEL * EDIM * 2);           //  4 MB

    const int M = BATCH * LSEQ;               // 2048 rows
    hipMemcpyAsync(h, x, (size_t)M * DMODEL * sizeof(float),
                   hipMemcpyDeviceToDevice, stream);

    for (int i = 0; i < NLAYER; ++i) {
        const float* in_w_i   = in_w   + (size_t)i * 2 * EDIM * DMODEL;
        const float* conv_w_i = conv_w + (size_t)i * EDIM * KCONV;
        const float* conv_b_i = conv_b + (size_t)i * EDIM;
        const float* xp_w_i   = xp_w   + (size_t)i * DBCW * EDIM;
        const float* dt_w_i   = dt_w   + (size_t)i * EDIM * RRANK;
        const float* dt_b_i   = dt_b   + (size_t)i * EDIM;
        const float* A_log_i  = A_log  + (size_t)i * EDIM * NSTATE;
        const float* D_p_i    = D_p    + (size_t)i * EDIM;
        const float* out_w_i  = out_w  + (size_t)i * DMODEL * EDIM;
        const float* norm_w_i = norm_w + (size_t)i * DMODEL;

        // 0) weight f16 copies (xp_w padded to 128 rows of zeros past N=96)
        cvt_f16_kernel<<<(2 * EDIM * DMODEL) / 256, 256, 0, stream>>>(
            in_w_i, in_w_h, 2 * EDIM * DMODEL, 2 * EDIM * DMODEL);
        cvt_f16_kernel<<<(BN * EDIM) / 256, 256, 0, stream>>>(
            xp_w_i, xp_w_h, DBCW * EDIM, BN * EDIM);
        cvt_f16_kernel<<<(EDIM * RRANK) / 256, 256, 0, stream>>>(
            dt_w_i, dt_w_h, EDIM * RRANK, EDIM * RRANK);
        cvt_f16_kernel<<<(DMODEL * EDIM) / 256, 256, 0, stream>>>(
            out_w_i, out_w_h, DMODEL * EDIM, DMODEL * EDIM);

        // 1) xn_h = f16(rmsnorm(h) * w)
        rmsnorm_kernel<<<M, 256, 0, stream>>>(h, norm_w_i, xn_h);

        // 2) xz = xn @ in_w^T        [2048,1024] x [4096,1024]^T
        gemm_wmma<0><<<dim3(2 * EDIM / BN, M / BM), 256, 0, stream>>>(
            xn_h, DMODEL, in_w_h, DMODEL, xz, 2 * EDIM, nullptr, nullptr,
            2 * EDIM, DMODEL);

        // 3) u/u_h = silu(causal_conv(xi) + cb)
        conv_silu_kernel<<<(M * EDIM) / 256, 256, 0, stream>>>(
            xz, conv_w_i, conv_b_i, u, u_h);

        // 4) dbc = u @ xp_w^T (f32 + f16 shadow)   [2048,2048] x [96,2048]^T
        gemm_wmma<3><<<dim3(1, M / BM), 256, 0, stream>>>(
            u_h, EDIM, xp_w_h, EDIM, dbc, DBCW, nullptr, dbc_h, DBCW, EDIM);

        // 5) delta = softplus(d_r @ dt_w^T + dt_b)  (d_r = dbc_h[:, :64], lda=96)
        gemm_wmma<1><<<dim3(EDIM / BN, M / BM), 256, 0, stream>>>(
            dbc_h, DBCW, dt_w_h, RRANK, delta, EDIM, dt_b_i, nullptr,
            EDIM, RRANK);

        // 6) y_h = f16(selective_scan(...) + D*u, gated by silu(z))
        scan_kernel<<<BATCH * (EDIM / 256), 256, 0, stream>>>(
            delta, u, xz, dbc, A_log_i, D_p_i, y_h);

        // 7) h = y @ out_w^T + h     [2048,2048] x [1024,2048]^T, residual add
        gemm_wmma<2><<<dim3(DMODEL / BN, M / BM), 256, 0, stream>>>(
            y_h, EDIM, out_w_h, EDIM, h, DMODEL, h, nullptr, DMODEL, EDIM);
    }
}
